// ToxicityLSTM_68212670595630
// MI455X (gfx1250) — compile-verified, matrix operand order checked by reading, not run on hardware
//
#include <hip/hip_runtime.h>
#include <hip/hip_bf16.h>

// ---------------------------------------------------------------------------
// ToxicityLSTM on MI455X (gfx1250, wave32, WMMA).
//
//  * Backward direction collapses to a single LSTM step per layer (the
//    re-reversed output at index T-1 is the backward scan's step 0).
//  * Recurrent kernel: 1 workgroup, 32 waves. g = [x_t ; h_{t-1}] @ [Wx;Wh]
//    (K=512), bf16 inputs, fp32 accum via v_wmma_f32_16x16x32_bf16.
//    Wave w handles M-tile (w>>4) and hidden tile (w&15): its 4 output tiles
//    are the i/j/f/o gate tiles of the SAME hidden units, so the LSTM cell
//    update is lane-local; c-state stays in 8 VGPRs/lane for all 512 steps.
//  * h_{t-1} lives in LDS (bf16, 16 KB); weights are streamed from the
//    L2-resident transposed bf16 matrix every step. An empty asm memory
//    clobber per step stops LICM from hoisting the (loop-invariant) weight
//    fragments into registers and spilling them to scratch.
// ---------------------------------------------------------------------------

typedef __attribute__((ext_vector_type(16))) __bf16 v16bf;
typedef __attribute__((ext_vector_type(8)))  float  v8f;

union BFrag { v16bf v; uint4 q[2]; };
union CFrag { v8f   v; float f[8]; };

__device__ __forceinline__ unsigned short f2bf(float f) {
  unsigned int u = __builtin_bit_cast(unsigned int, f);
  unsigned int r = (u + 0x7FFFu + ((u >> 16) & 1u)) >> 16;   // round-nearest-even
  return (unsigned short)r;
}
__device__ __forceinline__ float bf2f(unsigned short h) {
  unsigned int u = ((unsigned int)h) << 16;
  return __builtin_bit_cast(float, u);
}
__device__ __forceinline__ float sigf(float x) { return 1.0f / (1.0f + __expf(-x)); }
__device__ __forceinline__ float tanh_fast(float x) {
  return 1.0f - 2.0f / (__expf(2.0f * x) + 1.0f);   // one v_exp_f32 + rcp
}

// --------------------------------------------------------------------------
// Weight prep: WT[mat][n][k] (bf16, row-major over k=0..511) with
// k<256 -> Wx[l][k][n], k>=256 -> Wh[l][k-256][n]. mats: 0,1 = fw l0,l1;
// 2,3 = bw l0,l1.  4 * 1024 * 512 elements total.
// --------------------------------------------------------------------------
__global__ void prep_weights_kernel(const float* __restrict__ Wx_fw,
                                    const float* __restrict__ Wh_fw,
                                    const float* __restrict__ Wx_bw,
                                    const float* __restrict__ Wh_bw,
                                    unsigned short* __restrict__ WT) {
  int idx = blockIdx.x * 256 + threadIdx.x;          // < 4*1024*512
  int mat = idx >> 19;
  int rem = idx & ((1 << 19) - 1);
  int n   = rem >> 9;
  int k   = rem & 511;
  const float* Wx = (mat < 2) ? Wx_fw : Wx_bw;
  const float* Wh = (mat < 2) ? Wh_fw : Wh_bw;
  int l = mat & 1;
  float v = (k < 256)
      ? Wx[(size_t)l * 256 * 1024 + (size_t)k * 1024 + n]
      : Wh[(size_t)l * 256 * 1024 + (size_t)(k - 256) * 1024 + n];
  WT[(size_t)mat * 1024 * 512 + (size_t)n * 512 + k] = f2bf(v);
}

// --------------------------------------------------------------------------
// Embedding gather -> bf16, laid out [T][B][E] so each time step's A-matrix
// rows are contiguous.
// --------------------------------------------------------------------------
__global__ void embed_kernel(const int* __restrict__ tokens,
                             const float* __restrict__ emb,
                             unsigned short* __restrict__ xbf) {
  int bt = blockIdx.x;              // b*512 + t
  int e  = threadIdx.x;             // 0..255
  int b  = bt >> 9;
  int t  = bt & 511;
  int tok = tokens[b * 512 + t];
  xbf[((size_t)t * 32 + b) * 256 + e] = f2bf(emb[(size_t)tok * 256 + e]);
}

// --------------------------------------------------------------------------
// Fused LSTM layer: sequential over T steps, one workgroup (32 waves).
//   xin  : [T][32][256] bf16 (layer input sequence)
//   WT   : [1024][512]  bf16 (gate-major columns, K = [x ; h])
//   bias : [1024] fp32
//   hseq : [T][32][256] bf16 (layer output sequence)
// --------------------------------------------------------------------------
__global__ __launch_bounds__(1024) void lstm_layer_kernel(
    const unsigned short* __restrict__ xin,
    const unsigned short* __restrict__ WT,
    const float* __restrict__ bias,
    unsigned short* __restrict__ hseq,
    int T) {
  __shared__ unsigned short hsh[32 * 256];   // h_{t-1}, bf16, 16 KB

  const int tid   = threadIdx.x;
  const int wave  = tid >> 5;
  const int lane  = tid & 31;
  const int m     = wave >> 4;     // M tile: batch rows m*16 .. m*16+15
  const int hn    = wave & 15;     // hidden-unit tile: cols hn*16 .. hn*16+15
  const int row   = lane & 15;     // A-matrix row within M tile
  const int khalf = lane >> 4;     // lane-half K split (ISA A-layout)
  const int ncol  = hn * 16 + (lane & 15);   // hidden column for C/D layout
  const int hrow  = (m * 16 + row) * 256;    // LDS row base for A loads

  // zero-init h0
  for (int i = tid; i < 32 * 256; i += 1024) hsh[i] = 0;
  __syncthreads();

  // gate-column weight bases (i, j, f, o)
  const unsigned short* wb[4];
#pragma unroll
  for (int g = 0; g < 4; ++g)
    wb[g] = WT + (size_t)(g * 256 + ncol) * 512;

  float bv[4];
#pragma unroll
  for (int g = 0; g < 4; ++g) bv[g] = bias[g * 256 + ncol];

  CFrag cst;                        // fp32 cell state, register-resident
#pragma unroll
  for (int r = 0; r < 8; ++r) cst.f[r] = 0.0f;

  for (int t = 0; t < T; ++t) {
    // Compiler memory barrier: keep the (otherwise loop-invariant) weight
    // fragment loads inside the loop, streaming from L2 each step, instead
    // of LICM-hoisting 512 VGPRs of weights and spilling them to scratch.
    asm volatile("" ::: "memory");

    CFrag acc[4];
#pragma unroll
    for (int g = 0; g < 4; ++g)
#pragma unroll
      for (int r = 0; r < 8; ++r) acc[g].f[r] = bv[g];

    // ---- K = 0..255 : x_t contribution (A from global, L2-resident x) ----
    const unsigned short* xrowp = xin + ((size_t)t * 32 + m * 16 + row) * 256;
#pragma unroll
    for (int kb = 0; kb < 256; kb += 32) {
      const int k0 = kb + khalf * 8;
      BFrag a;
      a.q[0] = *(const uint4*)(xrowp + k0);
      a.q[1] = *(const uint4*)(xrowp + k0 + 16);
#pragma unroll
      for (int g = 0; g < 4; ++g) {
        BFrag b;
        const unsigned short* wp = wb[g] + k0;
        b.q[0] = *(const uint4*)(wp);
        b.q[1] = *(const uint4*)(wp + 16);
        acc[g].v = __builtin_amdgcn_wmma_f32_16x16x32_bf16(
            false, a.v, false, b.v, (short)0, acc[g].v, false, false);
      }
    }

    // ---- K = 256..511 : h_{t-1} contribution (A from LDS) ----
#pragma unroll
    for (int kb = 0; kb < 256; kb += 32) {
      const int k0 = kb + khalf * 8;
      BFrag a;
      a.q[0] = *(const uint4*)(&hsh[hrow + k0]);
      a.q[1] = *(const uint4*)(&hsh[hrow + k0 + 16]);
#pragma unroll
      for (int g = 0; g < 4; ++g) {
        BFrag b;
        const unsigned short* wp = wb[g] + 256 + k0;
        b.q[0] = *(const uint4*)(wp);
        b.q[1] = *(const uint4*)(wp + 16);
        acc[g].v = __builtin_amdgcn_wmma_f32_16x16x32_bf16(
            false, a.v, false, b.v, (short)0, acc[g].v, false, false);
      }
    }

    __syncthreads();   // everyone done reading h_{t-1} from LDS

    // ---- lane-local LSTM cell update (C-layout: VGPR r -> M = r(+8)) ----
#pragma unroll
    for (int r = 0; r < 8; ++r) {
      float gi = acc[0].f[r], gj = acc[1].f[r];
      float gf = acc[2].f[r], go = acc[3].f[r];
      float cn = sigf(gf + 1.0f) * cst.f[r] + sigf(gi) * tanh_fast(gj);
      float hv = sigf(go) * tanh_fast(cn);
      cst.f[r] = cn;
      const int M = m * 16 + r + khalf * 8;
      const unsigned short hb = f2bf(hv);
      hsh[M * 256 + ncol] = hb;
      hseq[((size_t)t * 32 + M) * 256 + ncol] = hb;
    }

    __syncthreads();   // h_t visible before next step's A loads

    if (t + 1 < T)     // gfx1250 global_prefetch_b8 for next step's x rows
      __builtin_prefetch(xin + ((size_t)(t + 1) * 32 + m * 16 + row) * 256, 0, 0);
  }
}

// --------------------------------------------------------------------------
// Head: logits = [h_fw ; h_bw] @ W_out + b_out ; sigmoid ; threshold.
// d_out = [logits(192) | output(192) | prediction(192)] fp32.
// --------------------------------------------------------------------------
__global__ void head_kernel(const unsigned short* __restrict__ hfw,
                            const unsigned short* __restrict__ hbw,
                            const float* __restrict__ Wout,
                            const float* __restrict__ bout,
                            float* __restrict__ out) {
  int tid = threadIdx.x;
  if (tid >= 192) return;
  int b = tid / 6, c = tid % 6;
  float s = bout[c];
  for (int k = 0; k < 256; ++k) s += bf2f(hfw[b * 256 + k]) * Wout[k * 6 + c];
  for (int k = 0; k < 256; ++k) s += bf2f(hbw[b * 256 + k]) * Wout[(256 + k) * 6 + c];
  out[tid] = s;
  float o = sigf(s);
  out[192 + tid] = o;
  out[384 + tid] = (o > 0.5f) ? 1.0f : 0.0f;
}

// --------------------------------------------------------------------------
extern "C" void kernel_launch(void* const* d_in, const int* in_sizes, int n_in,
                              void* d_out, int out_size, void* d_ws, size_t ws_size,
                              hipStream_t stream) {
  (void)in_sizes; (void)n_in; (void)out_size; (void)ws_size;

  const int*   tokens = (const int*)d_in[0];
  const float* emb    = (const float*)d_in[1];
  const float* Wx_fw  = (const float*)d_in[2];
  const float* Wh_fw  = (const float*)d_in[3];
  const float* b_fw   = (const float*)d_in[4];
  const float* Wx_bw  = (const float*)d_in[5];
  const float* Wh_bw  = (const float*)d_in[6];
  const float* b_bw   = (const float*)d_in[7];
  const float* W_out  = (const float*)d_in[8];
  const float* b_out  = (const float*)d_in[9];

  char* ws = (char*)d_ws;
  // workspace layout (bytes)
  unsigned short* xbf  = (unsigned short*)(ws);                         //  8 MiB  [T][32][256]
  unsigned short* WT   = (unsigned short*)(ws + (8ull << 20));          //  4 MiB  [4][1024][512]
  unsigned short* h0s  = (unsigned short*)(ws + (12ull << 20));         //  8 MiB  fw layer0 seq
  unsigned short* h1s  = (unsigned short*)(ws + (20ull << 20));         //  8 MiB  fw layer1 seq
  unsigned short* hb0  = (unsigned short*)(ws + (28ull << 20));         // 16 KiB  bw layer0 (T=1)
  unsigned short* hb1  = (unsigned short*)(ws + (28ull << 20) + (16ull << 10)); // 16 KiB

  const size_t MAT = (size_t)1024 * 512;

  // 1) weight transpose/concat/convert to bf16
  prep_weights_kernel<<<(4 * 1024 * 512) / 256, 256, 0, stream>>>(
      Wx_fw, Wh_fw, Wx_bw, Wh_bw, WT);

  // 2) embedding gather -> bf16 [T][B][E]
  embed_kernel<<<32 * 512, 256, 0, stream>>>(tokens, emb, xbf);

  // 3) forward stacked LSTM (full 512-step scans)
  lstm_layer_kernel<<<1, 1024, 0, stream>>>(xbf,           WT + 0 * MAT, b_fw,        h0s, 512);
  lstm_layer_kernel<<<1, 1024, 0, stream>>>(h0s,           WT + 1 * MAT, b_fw + 1024, h1s, 512);

  // 4) backward direction: re-reversed output at T-1 == backward step 0,
  //    which only needs x[:, T-1]; run the same kernel with T = 1.
  lstm_layer_kernel<<<1, 1024, 0, stream>>>(xbf + (size_t)511 * 32 * 256,
                                            WT + 2 * MAT, b_bw,        hb0, 1);
  lstm_layer_kernel<<<1, 1024, 0, stream>>>(hb0,           WT + 3 * MAT, b_bw + 1024, hb1, 1);

  // 5) head: logits / sigmoid / prediction
  head_kernel<<<1, 256, 0, stream>>>(h1s + (size_t)511 * 32 * 256, hb1,
                                     W_out, b_out, (float*)d_out);
}
